// MultiMamba_77884936946212
// MI455X (gfx1250) — matrix operand order repaired and checked by reading, not compiled, not run on hardware
//
#include <hip/hip_runtime.h>
#include <hip/hip_bf16.h>

// ---------------------------------------------------------------------------
// CDNA5 (gfx1250) implementation of the MultiMamba reference.
// GEMMs: v_wmma_f32_16x16x32_bf16, wave tile 32x64 (8 WMMA / K-step),
// A tile staged via global_load_async_to_lds_b128 with double buffering,
// weights pre-packed into a contiguous-k (k = 16*half + e) B-fragment layout (pure b128 loads).
// ---------------------------------------------------------------------------

typedef __bf16 v16bf __attribute__((ext_vector_type(16)));
typedef float  v8f   __attribute__((ext_vector_type(8)));

#define Dm    1024
#define Nn    2048
#define DSs   16
#define DCc   4
#define DIi   2048
#define DTRr  64
#define DBCp  128   // dbc columns padded 96 -> 128

__device__ __forceinline__ float silu_f(float v) {
    return v * (1.0f / (1.0f + __expf(-v)));
}
__device__ __forceinline__ float softplus_f(float v) {
    return (v > 20.0f) ? v : __logf(1.0f + __expf(v));
}

// ---------------------------------------------------------------------------
// Pack W[O x K] (f32, row-major) into WMMA B-fragment order (bf16):
//   P[(((nt*(K/32) + kc)*32 + lane)*16 + e] = W[nt*16 + (lane&15)]
//                                              [kc*32 + e + (lane>>4)*16]
// Zero-fills o >= O (for the 96 -> 128 pad).
// ---------------------------------------------------------------------------
__global__ __launch_bounds__(256) void pack_b_kernel(
    const float* __restrict__ W, __bf16* __restrict__ P,
    int O, int Opad, int K) {
    int idx = blockIdx.x * 256 + threadIdx.x;
    if (idx >= Opad * K) return;
    int e    = idx & 15;
    int l    = (idx >> 4) & 31;
    int rest = idx >> 9;
    int kch  = K >> 5;
    int kc   = rest % kch;
    int nt   = rest / kch;
    int o = nt * 16 + (l & 15);
    int k = kc * 32 + e + (l >> 4) * 16;
    P[idx] = (o < O) ? (__bf16)W[(size_t)o * K + k] : (__bf16)0.0f;
}

// ---------------------------------------------------------------------------
// LayerNorm over a (possibly concatenated) row, bf16 output for WMMA input.
// ---------------------------------------------------------------------------
__global__ __launch_bounds__(256) void ln_bf16_kernel(
    const float* __restrict__ in0, const float* __restrict__ in1,
    int d0, int d1,
    const float* __restrict__ w, const float* __restrict__ b,
    __bf16* __restrict__ out) {
    const int row = blockIdx.x;
    const int d   = d0 + d1;
    __shared__ float r1[256];
    __shared__ float r2[256];
    float s = 0.f, sq = 0.f;
    for (int i = threadIdx.x; i < d; i += 256) {
        float v = (i < d0) ? in0[(size_t)row * d0 + i]
                           : in1[(size_t)row * d1 + (i - d0)];
        s += v; sq += v * v;
    }
    r1[threadIdx.x] = s; r2[threadIdx.x] = sq;
    __syncthreads();
    for (int st = 128; st > 0; st >>= 1) {
        if (threadIdx.x < st) {
            r1[threadIdx.x] += r1[threadIdx.x + st];
            r2[threadIdx.x] += r2[threadIdx.x + st];
        }
        __syncthreads();
    }
    const float mu   = r1[0] / d;
    const float var  = r2[0] / d - mu * mu;
    const float rstd = rsqrtf(var + 1e-5f);
    for (int i = threadIdx.x; i < d; i += 256) {
        float v = (i < d0) ? in0[(size_t)row * d0 + i]
                           : in1[(size_t)row * d1 + (i - d0)];
        out[(size_t)row * d + i] = (__bf16)((v - mu) * rstd * w[i] + b[i]);
    }
}

// ---------------------------------------------------------------------------
// Tiled bf16 WMMA GEMM:  C[M x N] = A[M x K] * B[K x N]  (+bias, +act)
// Block: 256 threads = 8 waves (4 M-waves x 2 N-waves), block tile 128x128,
// wave tile 32x64 -> 8 v_wmma per K-step.  B pre-packed in fragment order
// (global b128 loads); A tile async-DMA'd to double-buffered LDS.
//   act: 0 none, 1 silu, 2 softplus.  out_bf16: bf16 vs f32 store.
// ---------------------------------------------------------------------------
#define ASTR 40   // LDS row stride (elements): 32 data + 8 pad (de-stagger banks)

__device__ __forceinline__ void stage_async(
    const __bf16* __restrict__ A, int bm, int K, int kcx,
    __bf16* buf, int tid) {
    // 128 rows x 32 cols; each thread DMAs a 32B row-chunk (2 x b128).
    const int r  = tid >> 1;        // 0..127
    const int cg = (tid & 1) * 2;   // 0 or 2 (16B units)
    const __bf16* g = A + (size_t)(bm + r) * K + kcx * 32 + cg * 8;
    const unsigned int l = (unsigned int)(unsigned long long)(buf + r * ASTR + cg * 8);
    const unsigned long long ga = (unsigned long long)g;
    asm volatile("global_load_async_to_lds_b128 %0, %1, off"
                 :: "v"(l), "v"(ga) : "memory");
    asm volatile("global_load_async_to_lds_b128 %0, %1, off offset:16"
                 :: "v"(l), "v"(ga) : "memory");
}

__global__ __launch_bounds__(256) void gemm_bf16_wmma_kernel(
    const __bf16* __restrict__ A, const __bf16* __restrict__ Bpk,
    const float* __restrict__ bias, void* __restrict__ Cp,
    int M, int N, int K, int act, int out_bf16) {
    __shared__ __bf16 sA[2][128 * ASTR];

    const int tid  = threadIdx.x;
    const int lane = tid & 31;
    const int wave = tid >> 5;
    const int wm   = wave & 3;   // 0..3 : 32-row slice
    const int wn   = wave >> 2;  // 0..1 : 64-col slice
    const int bm   = blockIdx.y * 128;
    const int bn   = blockIdx.x * 128;
    const int half = lane >> 4;
    const int lr   = lane & 15;
    const int kch  = K >> 5;
    const int nt0  = (bn >> 4) + wn * 4;   // first of four 16-col tiles

    v8f acc[2][4] = {};
    union U { v16bf v; uint4 q[2]; };

    stage_async(A, bm, K, 0, sA[0], tid);

    for (int kc = 0; kc < kch; ++kc) {
        __bf16* cur = sA[kc & 1];
        if (kc + 1 < kch) {
            stage_async(A, bm, K, kc + 1, sA[(kc + 1) & 1], tid);
            asm volatile("s_wait_asynccnt 0x2" ::: "memory");
        } else {
            asm volatile("s_wait_asynccnt 0x0" ::: "memory");
        }
        __syncthreads();

        // ---- B fragments straight from packed global (2 x b128 each) ----
        U b[4];
        #pragma unroll
        for (int nt = 0; nt < 4; ++nt) {
            const uint4* p = reinterpret_cast<const uint4*>(
                Bpk + (((size_t)(nt0 + nt) * kch + kc) * 32 + lane) * 16);
            b[nt].q[0] = p[0];
            b[nt].q[1] = p[1];
        }
        // ---- A fragments from LDS (2 x ds_load_b128 each) ----
        U a[2];
        #pragma unroll
        for (int mt = 0; mt < 2; ++mt) {
            const __bf16* ar = cur + (wm * 32 + mt * 16 + lr) * ASTR;
            a[mt].q[0] = *(reinterpret_cast<const uint4*>(ar) + half * 2);
            a[mt].q[1] = *(reinterpret_cast<const uint4*>(ar) + half * 2 + 1);
        }

        #pragma unroll
        for (int mt = 0; mt < 2; ++mt)
            #pragma unroll
            for (int nt = 0; nt < 4; ++nt)
                acc[mt][nt] = __builtin_amdgcn_wmma_f32_16x16x32_bf16(
                    false, a[mt].v, false, b[nt].v, (short)0, acc[mt][nt],
                    false, false);
        __syncthreads();
    }

    // ---- epilogue: C/D layout: VGPR r -> M = r + half*8, N = lr ----
    #pragma unroll
    for (int mt = 0; mt < 2; ++mt) {
        #pragma unroll
        for (int nt = 0; nt < 4; ++nt) {
            const int col = bn + wn * 64 + nt * 16 + lr;
            const float bv = bias ? bias[col] : 0.0f;
            #pragma unroll
            for (int r = 0; r < 8; ++r) {
                const int row = bm + wm * 32 + mt * 16 + r + half * 8;
                float v = acc[mt][nt][r] + bv;
                if (act == 1)      v = silu_f(v);
                else if (act == 2) v = softplus_f(v);
                if (out_bf16)
                    ((__bf16*)Cp)[(size_t)row * N + col] = (__bf16)v;
                else
                    ((float*)Cp)[(size_t)row * N + col] = v;
            }
        }
    }
}

// ---------------------------------------------------------------------------
// Causal depthwise conv (k=4) over sequence + SiLU. xc = xz[:, 0:DI] (bf16).
// ---------------------------------------------------------------------------
__global__ __launch_bounds__(256) void conv_silu_kernel(
    const __bf16* __restrict__ xz, const float* __restrict__ conv_w,
    const float* __restrict__ conv_b, float* __restrict__ u,
    __bf16* __restrict__ u_bf) {
    int idx = blockIdx.x * 256 + threadIdx.x;  // t * DI + c
    int t = idx / DIi;
    int c = idx % DIi;
    float acc = conv_b[c];
    #pragma unroll
    for (int k = 0; k < DCc; ++k) {
        int tt = t + k - (DCc - 1);
        if (tt >= 0)
            acc += conv_w[c * DCc + k] * (float)xz[(size_t)tt * (2 * DIi) + c];
    }
    float s = silu_f(acc);
    u[idx]    = s;
    u_bf[idx] = (__bf16)s;
}

// dbc[:, 0:64] -> contiguous bf16 dt matrix (dbc row stride = DBCp)
__global__ __launch_bounds__(256) void dt_extract_kernel(
    const float* __restrict__ dbc, __bf16* __restrict__ dt_bf) {
    int idx = blockIdx.x * 256 + threadIdx.x;  // t * 64 + j
    int t = idx >> 6;
    int j = idx & 63;
    dt_bf[idx] = (__bf16)dbc[(size_t)t * DBCp + j];
}

// ---------------------------------------------------------------------------
// Selective scan: one lane per channel, sequential over t, B/C in LDS.
// Fuses D-skip and silu(z) gating; writes y as bf16.
// ---------------------------------------------------------------------------
__global__ __launch_bounds__(256) void scan_kernel(
    const float* __restrict__ u, const float* __restrict__ delta,
    const float* __restrict__ dbc, const __bf16* __restrict__ xz,
    const float* __restrict__ A_log, const float* __restrict__ Dp,
    __bf16* __restrict__ y_bf) {
    const int c = blockIdx.x * 256 + threadIdx.x;
    float A[DSs];
    #pragma unroll
    for (int s = 0; s < DSs; ++s) A[s] = -__expf(A_log[(size_t)c * DSs + s]);
    float h[DSs];
    #pragma unroll
    for (int s = 0; s < DSs; ++s) h[s] = 0.f;
    const float dpc = Dp[c];

    __shared__ float sBC[32];  // [0:16) = B_t, [16:32) = C_t
    for (int t = 0; t < Nn; ++t) {
        __syncthreads();
        if (threadIdx.x < 32)
            sBC[threadIdx.x] = dbc[(size_t)t * DBCp + DTRr + threadIdx.x];
        __syncthreads();
        const float dt_ = delta[(size_t)t * DIi + c];
        const float ut  = u[(size_t)t * DIi + c];
        const float du  = dt_ * ut;
        float acc = 0.f;
        #pragma unroll
        for (int s = 0; s < DSs; ++s) {
            float dA = __expf(dt_ * A[s]);
            h[s] = h[s] * dA + du * sBC[s];
            acc += h[s] * sBC[16 + s];
        }
        const float z = (float)xz[(size_t)t * (2 * DIi) + DIi + c];
        const float yv = (acc + ut * dpc) * silu_f(z);
        y_bf[(size_t)t * DIi + c] = (__bf16)yv;
    }
}

// ---------------------------------------------------------------------------
// Final inverse-permutation scatter (argsort of a permutation = inverse).
// ---------------------------------------------------------------------------
__global__ __launch_bounds__(256) void scatter_small_kernel(
    const float* __restrict__ coords, const int* __restrict__ comps,
    const int* __restrict__ indices, float* __restrict__ out) {
    int i = blockIdx.x * 256 + threadIdx.x;
    if (i >= Nn) return;
    int dst = indices[i];
    out[dst * 3 + 0] = coords[i * 3 + 0];
    out[dst * 3 + 1] = coords[i * 3 + 1];
    out[dst * 3 + 2] = coords[i * 3 + 2];
    float* comps_out = out + (size_t)Nn * 3 + (size_t)Nn * Dm;
    comps_out[dst] = (float)comps[i];
}

__global__ __launch_bounds__(256) void scatter_nodes_kernel(
    const float* __restrict__ x, const int* __restrict__ indices,
    float* __restrict__ out) {
    int idx = blockIdx.x * 256 + threadIdx.x;  // i * (D/4) + j4
    int i  = idx / (Dm / 4);
    int j4 = idx % (Dm / 4);
    float4* nodes_out = reinterpret_cast<float4*>(out + (size_t)Nn * 3);
    const float4* src = reinterpret_cast<const float4*>(x);
    nodes_out[(size_t)indices[i] * (Dm / 4) + j4] = src[(size_t)i * (Dm / 4) + j4];
}

// ---------------------------------------------------------------------------
// Host launcher
// ---------------------------------------------------------------------------
static inline size_t align256(size_t x) { return (x + 255) & ~(size_t)255; }

extern "C" void kernel_launch(void* const* d_in, const int* in_sizes, int n_in,
                              void* d_out, int out_size, void* d_ws, size_t ws_size,
                              hipStream_t stream) {
    (void)in_sizes; (void)n_in; (void)out_size; (void)ws_size;

    const float* coords    = (const float*)d_in[0];
    const float* nodes     = (const float*)d_in[1];
    const int*   comps     = (const int*)d_in[2];
    const int*   indices   = (const int*)d_in[3];
    const float* ln_w      = (const float*)d_in[4];
    const float* ln_b      = (const float*)d_in[5];
    const float* in_proj_w = (const float*)d_in[6];
    const float* conv_w    = (const float*)d_in[7];
    const float* conv_b    = (const float*)d_in[8];
    const float* x_proj_w  = (const float*)d_in[9];
    const float* dt_w      = (const float*)d_in[10];
    const float* dt_b      = (const float*)d_in[11];
    const float* A_log     = (const float*)d_in[12];
    const float* D_param   = (const float*)d_in[13];
    const float* out_proj_w= (const float*)d_in[14];
    const float* cl_ln1_w  = (const float*)d_in[15];
    const float* cl_ln1_b  = (const float*)d_in[16];
    const float* cl_w1     = (const float*)d_in[17];
    const float* cl_b1     = (const float*)d_in[18];
    const float* cl_ln2_w  = (const float*)d_in[19];
    const float* cl_ln2_b  = (const float*)d_in[20];
    const float* cl_w2     = (const float*)d_in[21];
    const float* cl_b2     = (const float*)d_in[22];
    float* out = (float*)d_out;

    // ---- workspace layout ----
    char* ws = (char*)d_ws;
    size_t off = 0;
    auto alloc = [&](size_t bytes) -> void* {
        void* p = ws + off;
        off = align256(off + bytes);
        return p;
    };
    __bf16* w_in  = (__bf16*)alloc((size_t)4096 * Dm * 2);       // packed
    __bf16* w_xp  = (__bf16*)alloc((size_t)DBCp * DIi * 2);      // packed (padded)
    __bf16* w_dt  = (__bf16*)alloc((size_t)DIi * DTRr * 2);      // packed
    __bf16* w_out = (__bf16*)alloc((size_t)Dm * DIi * 2);        // packed
    __bf16* w_c1  = (__bf16*)alloc((size_t)(2*Dm) * (2*Dm) * 2); // packed
    __bf16* w_c2  = (__bf16*)alloc((size_t)Dm * (2*Dm) * 2);     // packed
    float*  xbuf  = (float*)alloc((size_t)Nn * Dm * 4);
    __bf16* ab    = (__bf16*)alloc((size_t)Nn * (2*Dm) * 2);     // shared bf16 staging
    __bf16* xz    = (__bf16*)alloc((size_t)Nn * (2*DIi) * 2);
    float*  u     = (float*)alloc((size_t)Nn * DIi * 4);
    float*  dbc   = (float*)alloc((size_t)Nn * DBCp * 4);
    __bf16* dt_bf = (__bf16*)alloc((size_t)Nn * DTRr * 2);
    float*  delta = (float*)alloc((size_t)Nn * DIi * 4);
    float*  upd   = (float*)alloc((size_t)Nn * Dm * 4);
    float*  hbuf  = (float*)alloc((size_t)Nn * (2*Dm) * 4);

    // ---- one-time weight packing to WMMA B-fragment layout (bf16) ----
    pack_b_kernel<<<(4096*Dm)/256,      256, 0, stream>>>(in_proj_w,  w_in,  4096, 4096, Dm);
    pack_b_kernel<<<(DBCp*DIi)/256,     256, 0, stream>>>(x_proj_w,   w_xp,  96,   DBCp, DIi);
    pack_b_kernel<<<(DIi*DTRr)/256,     256, 0, stream>>>(dt_w,       w_dt,  DIi,  DIi,  DTRr);
    pack_b_kernel<<<(Dm*DIi)/256,       256, 0, stream>>>(out_proj_w, w_out, Dm,   Dm,   DIi);
    pack_b_kernel<<<(2*Dm*2*Dm)/256,    256, 0, stream>>>(cl_w1,      w_c1,  2*Dm, 2*Dm, 2*Dm);
    pack_b_kernel<<<(Dm*2*Dm)/256,      256, 0, stream>>>(cl_w2,      w_c2,  Dm,   Dm,   2*Dm);

    // x = nodes[0]
    hipMemcpyAsync(xbuf, nodes, (size_t)Nn * Dm * sizeof(float),
                   hipMemcpyDeviceToDevice, stream);

    for (int loop = 0; loop < 2; ++loop) {
        // -------- mamba --------
        ln_bf16_kernel<<<Nn, 256, 0, stream>>>(xbuf, nullptr, Dm, 0, ln_w, ln_b, ab);
        // xz = xn @ in_proj_w.T  (2048 x 4096, bf16 out)
        gemm_bf16_wmma_kernel<<<dim3(4096/128, Nn/128), 256, 0, stream>>>(
            ab, w_in, nullptr, xz, Nn, 4096, Dm, 0, 1);
        // u = silu(conv(xc))
        conv_silu_kernel<<<(Nn*DIi)/256, 256, 0, stream>>>(xz, conv_w, conv_b, u, ab);
        // dbc = u @ x_proj_w.T  (2048 x 128(padded), f32)
        gemm_bf16_wmma_kernel<<<dim3(DBCp/128, Nn/128), 256, 0, stream>>>(
            ab, w_xp, nullptr, dbc, Nn, DBCp, DIi, 0, 0);
        // dt slice -> bf16
        dt_extract_kernel<<<(Nn*DTRr)/256, 256, 0, stream>>>(dbc, dt_bf);
        // delta = softplus(dt @ dt_w.T + dt_b)  (2048 x 2048, f32)
        gemm_bf16_wmma_kernel<<<dim3(DIi/128, Nn/128), 256, 0, stream>>>(
            dt_bf, w_dt, dt_b, delta, Nn, DIi, DTRr, 2, 0);
        // selective scan (fused D-skip + z gating) -> y bf16 in ab
        scan_kernel<<<DIi/256, 256, 0, stream>>>(u, delta, dbc, xz, A_log, D_param, ab);
        // upd = y @ out_proj_w.T  (2048 x 1024, f32)
        gemm_bf16_wmma_kernel<<<dim3(Dm/128, Nn/128), 256, 0, stream>>>(
            ab, w_out, nullptr, upd, Nn, Dm, DIi, 0, 0);

        // -------- residual concat MLP --------
        ln_bf16_kernel<<<Nn, 256, 0, stream>>>(xbuf, upd, Dm, Dm, cl_ln1_w, cl_ln1_b, ab);
        gemm_bf16_wmma_kernel<<<dim3((2*Dm)/128, Nn/128), 256, 0, stream>>>(
            ab, w_c1, cl_b1, hbuf, Nn, 2*Dm, 2*Dm, 1, 0);
        ln_bf16_kernel<<<Nn, 256, 0, stream>>>(hbuf, nullptr, 2*Dm, 0, cl_ln2_w, cl_ln2_b, ab);
        gemm_bf16_wmma_kernel<<<dim3(Dm/128, Nn/128), 256, 0, stream>>>(
            ab, w_c2, cl_b2, xbuf, Nn, Dm, 2*Dm, 1, 0);
    }

    // -------- final inverse-permutation gather --------
    scatter_small_kernel<<<(Nn + 255)/256, 256, 0, stream>>>(coords, comps, indices, out);
    scatter_nodes_kernel<<<(Nn*(Dm/4))/256, 256, 0, stream>>>(xbuf, indices, out);
}
